// MambaModel_84834194031083
// MI455X (gfx1250) — matrix operand order repaired and checked by reading, not compiled, not run on hardware
//
#include <hip/hip_runtime.h>
#include <hip/hip_bf16.h>

typedef __attribute__((ext_vector_type(16))) _Float16 v16h;
typedef __attribute__((ext_vector_type(8)))  _Float16 v8h;
typedef __attribute__((ext_vector_type(8)))  float    v8f;
typedef __attribute__((ext_vector_type(4)))  unsigned int u32x4;
typedef __attribute__((ext_vector_type(8)))  int      i32x8;
typedef __attribute__((ext_vector_type(4)))  int      i32x4;

#define BB   2
#define TT   2048
#define VV   32000
#define DM   1024
#define DIc  2048
#define NNs  16
#define RRr  64
#define LLl  4
#define KKc  4
#define BT   (BB*TT)
#define PW   128          // padded width for x_proj outputs (96 -> 128)

// ---------------------------------------------------------------------------
// f32 -> f16 convert (grid-stride)
// ---------------------------------------------------------------------------
__global__ void cvt_f16_kernel(const float* __restrict__ src, _Float16* __restrict__ dst, size_t n) {
    for (size_t i = (size_t)blockIdx.x * blockDim.x + threadIdx.x; i < n;
         i += (size_t)gridDim.x * blockDim.x)
        dst[i] = (_Float16)src[i];
}

__global__ void zero_f16_kernel(_Float16* __restrict__ dst, size_t n) {
    for (size_t i = (size_t)blockIdx.x * blockDim.x + threadIdx.x; i < n;
         i += (size_t)gridDim.x * blockDim.x)
        dst[i] = (_Float16)0.f;
}

// ---------------------------------------------------------------------------
// Embedding gather
// ---------------------------------------------------------------------------
__global__ void embed_kernel(const int* __restrict__ ids, const float* __restrict__ W,
                             float* __restrict__ x) {
    const size_t n = (size_t)BT * DM;
    for (size_t i = (size_t)blockIdx.x * blockDim.x + threadIdx.x; i < n;
         i += (size_t)gridDim.x * blockDim.x) {
        int row = (int)(i / DM), d = (int)(i % DM);
        x[i] = W[(size_t)ids[row] * DM + d];
    }
}

// ---------------------------------------------------------------------------
// LayerNorm over DM=1024 -> f16 (one block / row)
// ---------------------------------------------------------------------------
__global__ __launch_bounds__(256)
void ln_f16_kernel(const float* __restrict__ x, const float* __restrict__ w,
                   const float* __restrict__ b, _Float16* __restrict__ out) {
    __shared__ float red[256];
    const int row = blockIdx.x;
    const float* xr = x + (size_t)row * DM;
    float s = 0.f;
    for (int d = threadIdx.x; d < DM; d += 256) s += xr[d];
    red[threadIdx.x] = s; __syncthreads();
    for (int o = 128; o > 0; o >>= 1) {
        if (threadIdx.x < o) red[threadIdx.x] += red[threadIdx.x + o];
        __syncthreads();
    }
    const float mu = red[0] * (1.f / DM);
    __syncthreads();
    float v = 0.f;
    for (int d = threadIdx.x; d < DM; d += 256) { float t = xr[d] - mu; v += t * t; }
    red[threadIdx.x] = v; __syncthreads();
    for (int o = 128; o > 0; o >>= 1) {
        if (threadIdx.x < o) red[threadIdx.x] += red[threadIdx.x + o];
        __syncthreads();
    }
    const float rstd = rsqrtf(red[0] * (1.f / DM) + 1e-5f);
    for (int d = threadIdx.x; d < DM; d += 256)
        out[(size_t)row * DM + d] = (_Float16)((xr[d] - mu) * rstd * w[d] + b[d]);
}

// ---------------------------------------------------------------------------
// TDM issue: 2D tile (tile_dim0=32 halfs along K, tile_dim1=128 rows) from
// global f16 matrix (row stride = lda elements) into LDS at lds_off.
// D# per CDNA5 ISA ch.8: group0 = count/lds_addr/global_addr/type,
// group1 = data_size + tensor dims + tile dims + strides.
// ---------------------------------------------------------------------------
__device__ __forceinline__
void tdm_load_tile(const _Float16* gptr, unsigned int lds_off,
                   unsigned int lda, unsigned int rows) {
    const unsigned long long ga = (unsigned long long)(uintptr_t)gptr;
    u32x4 g0;
    g0[0] = 1u;                                             // count=1, user mode
    g0[1] = lds_off;                                        // lds_addr (bytes)
    g0[2] = (unsigned int)ga;                               // global_addr[31:0]
    g0[3] = (unsigned int)((ga >> 32) & 0x01FFFFFFu)        // global_addr[56:32]
          | 0x80000000u;                                    // type=2 ("image")
    i32x8 g1;
    g1[0] = (int)(1u << 16);                                // data_size=1 -> 2 bytes
    g1[1] = (int)((lda & 0xFFFFu) << 16);                   // tensor_dim0[15:0]
    g1[2] = (int)((lda >> 16) | ((rows & 0xFFFFu) << 16));  // td0[31:16] | td1[15:0]
    g1[3] = (int)((rows >> 16) | (32u << 16));              // td1[31:16] | tile_dim0=32
    g1[4] = (int)128u;                                      // tile_dim1=128, tile_dim2=0
    g1[5] = (int)lda;                                       // tensor_dim0_stride[31:0]
    g1[6] = 0;                                              // stride0[47:32] | stride1[15:0]
    g1[7] = 0;                                              // stride1[47:16]
    i32x4 z4 = {};
    i32x8 z8 = {};
    __builtin_amdgcn_tensor_load_to_lds(g0, g1, z4, z4, z8, 0);
}

// ---------------------------------------------------------------------------
// WMMA GEMM: C[m,n] = sum_k A[m,k] * B[n,k]
//   A: MxK f16 (row stride lda), staged to LDS by the Tensor Data Mover
//   B: NxK f16 (row stride ldb) -- weight matrix, contiguous per-lane loads
//   C: f32 (row stride ldc)
// Requires M%128==0, N%64==0, K%32==0, lda/ldb mult. of 16 -> no guards.
// block = 128 thr = 4 waves along M; wave tile 32x64; block tile 128x64.
// ---------------------------------------------------------------------------
__global__ __launch_bounds__(128)
void gemm_wmma_f16(const _Float16* __restrict__ A, const _Float16* __restrict__ Bw,
                   float* __restrict__ C, int M, int N, int K,
                   int lda, int ldb, int ldc) {
    __shared__ __align__(16) _Float16 Atile[2][128 * 32];

    const int lane    = threadIdx.x & 31;
    const int wave    = threadIdx.x >> 5;
    const int m0      = blockIdx.x * 128;
    const int n0      = blockIdx.y * 64;
    const int halfSel = lane >> 4;
    const int l15     = lane & 15;

    const _Float16* ablk = A + (size_t)m0 * lda;

    // prologue: stage k0=0 tile
    if (wave == 0) {
        tdm_load_tile(ablk, (unsigned int)(uintptr_t)&Atile[0][0],
                      (unsigned int)lda, (unsigned int)M);
        __builtin_amdgcn_s_wait_tensorcnt(0);
    }
    __syncthreads();

    v8f acc[8];
#pragma unroll
    for (int j = 0; j < 8; ++j) acc[j] = {};

    const int r0 = wave * 32 + l15;          // local A row for frag 0
    const int kbB_off = halfSel * 8;         // B K-offset within step
    const int kbA_off = halfSel * 8;         // A K-offset within step

    for (int k0 = 0; k0 < K; k0 += 32) {
        const int cur = (k0 >> 5) & 1;
        if (wave == 0 && k0 + 32 < K)
            tdm_load_tile(ablk + (k0 + 32),
                          (unsigned int)(uintptr_t)&Atile[cur ^ 1][0],
                          (unsigned int)lda, (unsigned int)M);

        // ---- A fragments from LDS (documented 16-bit A layout)
        const _Float16* ap0 = &Atile[cur][(size_t)r0 * 32 + kbA_off];
        const _Float16* ap1 = ap0 + 16 * 32;
        v8h lo0 = *(const v8h*)ap0, hi0 = *(const v8h*)(ap0 + 16);
        v8h lo1 = *(const v8h*)ap1, hi1 = *(const v8h*)(ap1 + 16);
        v16h a0 = __builtin_shufflevector(lo0, hi0, 0,1,2,3,4,5,6,7,8,9,10,11,12,13,14,15);
        v16h a1 = __builtin_shufflevector(lo1, hi1, 0,1,2,3,4,5,6,7,8,9,10,11,12,13,14,15);

        // ---- B fragments from global (two 16B chunks per lane, table order)
        const int kbB = k0 + kbB_off;
        v16h bf[4];
#pragma unroll
        for (int j = 0; j < 4; ++j) {
            const int col = n0 + j * 16 + l15;
            const _Float16* bp = Bw + (size_t)col * ldb + kbB;
            v8h blo = *(const v8h*)bp, bhi = *(const v8h*)(bp + 16);
            bf[j] = __builtin_shufflevector(blo, bhi, 0,1,2,3,4,5,6,7,8,9,10,11,12,13,14,15);
        }
        __builtin_prefetch(Bw + (size_t)(n0 + l15) * ldb + kbB + 32, 0, 3);

#pragma unroll
        for (int j = 0; j < 4; ++j) {
            acc[j]     = __builtin_amdgcn_wmma_f32_16x16x32_f16(
                false, a0, false, bf[j], (short)0, acc[j],     false, false);
            acc[4 + j] = __builtin_amdgcn_wmma_f32_16x16x32_f16(
                false, a1, false, bf[j], (short)0, acc[4 + j], false, false);
        }

        if (wave == 0) __builtin_amdgcn_s_wait_tensorcnt(0);
        __syncthreads();
    }

    // ---- store D (f32 C/D layout: VGPR r -> M = 8*halfSel + r)
    const int rbase = m0 + wave * 32 + halfSel * 8;
#pragma unroll
    for (int j = 0; j < 4; ++j) {
        const int col = n0 + j * 16 + l15;
#pragma unroll
        for (int r = 0; r < 8; ++r) {
            C[(size_t)(rbase + r)      * ldc + col] = acc[j][r];
            C[(size_t)(rbase + 16 + r) * ldc + col] = acc[4 + j][r];
        }
    }
}

// ---------------------------------------------------------------------------
// Causal depthwise conv (K=4) + SiLU; input xz[:, :DI], row width 2*DI
// ---------------------------------------------------------------------------
__global__ void conv_silu_kernel(const float* __restrict__ xz, const float* __restrict__ w,
                                 const float* __restrict__ bias, float* __restrict__ xc,
                                 _Float16* __restrict__ xcf16) {
    const size_t n = (size_t)BT * DIc;
    for (size_t i = (size_t)blockIdx.x * blockDim.x + threadIdx.x; i < n;
         i += (size_t)gridDim.x * blockDim.x) {
        const int row = (int)(i / DIc), c = (int)(i % DIc);
        const int b = row / TT, t = row % TT;
        float s = bias[c];
#pragma unroll
        for (int k = 0; k < KKc; ++k) {
            const int tt = t - (KKc - 1) + k;
            if (tt >= 0)
                s += w[c * KKc + k] * xz[(size_t)(b * TT + tt) * (2 * DIc) + c];
        }
        const float sv = s / (1.f + __expf(-s));
        xc[i] = sv;
        xcf16[i] = (_Float16)sv;
    }
}

// ---------------------------------------------------------------------------
// dt epilogue: dt = softplus(g + dtb[c])
// ---------------------------------------------------------------------------
__global__ void dt_epilogue_kernel(const float* __restrict__ g, const float* __restrict__ dtb,
                                   float* __restrict__ dt) {
    const size_t n = (size_t)BT * DIc;
    for (size_t i = (size_t)blockIdx.x * blockDim.x + threadIdx.x; i < n;
         i += (size_t)gridDim.x * blockDim.x) {
        const int c = (int)(i % DIc);
        const float x = g[i] + dtb[c];
        dt[i] = (x > 20.f) ? x : log1pf(__expf(x));
    }
}

// ---------------------------------------------------------------------------
// Selective scan: one lane per (b, channel); sequential over T.
// proj row stride PW=128: [dt_r(64) | B(16) | C(16) | pad(32)]
// ---------------------------------------------------------------------------
__global__ __launch_bounds__(128)
void scan_kernel(const float* __restrict__ xc, const float* __restrict__ dt,
                 const float* __restrict__ proj, const float* __restrict__ A_log,
                 const float* __restrict__ Dp, float* __restrict__ y) {
    const int gc = blockIdx.x * 128 + threadIdx.x;   // 0 .. B*DI-1
    const int b = gc / DIc, c = gc % DIc;
    float Aneg[NNs], h[NNs];
#pragma unroll
    for (int nn = 0; nn < NNs; ++nn) {
        Aneg[nn] = -__expf(A_log[(size_t)c * NNs + nn]);
        h[nn] = 0.f;
    }
    const float Dc = Dp[c];
    for (int t = 0; t < TT; ++t) {
        const int row = b * TT + t;
        const float u = xc[(size_t)row * DIc + c];
        const float d = dt[(size_t)row * DIc + c];
        const float* pr = proj + (size_t)row * PW;
        const float du = d * u;
        float yv = u * Dc;
#pragma unroll
        for (int nn = 0; nn < NNs; ++nn) {
            const float dA = __expf(d * Aneg[nn]);
            h[nn] = dA * h[nn] + du * pr[RRr + nn];
            yv += h[nn] * pr[RRr + NNs + nn];
        }
        y[(size_t)row * DIc + c] = yv;
    }
}

// ---------------------------------------------------------------------------
// Gate: y *= silu(z), emit f16
// ---------------------------------------------------------------------------
__global__ void gate_kernel(float* __restrict__ y, const float* __restrict__ xz,
                            _Float16* __restrict__ yf16) {
    const size_t n = (size_t)BT * DIc;
    for (size_t i = (size_t)blockIdx.x * blockDim.x + threadIdx.x; i < n;
         i += (size_t)gridDim.x * blockDim.x) {
        const int row = (int)(i / DIc), c = (int)(i % DIc);
        const float z = xz[(size_t)row * (2 * DIc) + DIc + c];
        const float s = z / (1.f + __expf(-z));
        const float v = y[i] * s;
        y[i] = v;
        yf16[i] = (_Float16)v;
    }
}

__global__ void add_res_kernel(float* __restrict__ x, const float* __restrict__ t, size_t n) {
    for (size_t i = (size_t)blockIdx.x * blockDim.x + threadIdx.x; i < n;
         i += (size_t)gridDim.x * blockDim.x)
        x[i] += t[i];
}

__global__ void add_bias_kernel(float* __restrict__ out, const float* __restrict__ bias) {
    const size_t n = (size_t)BT * VV;
    for (size_t i = (size_t)blockIdx.x * blockDim.x + threadIdx.x; i < n;
         i += (size_t)gridDim.x * blockDim.x)
        out[i] += bias[i % VV];
}

// ---------------------------------------------------------------------------
// Host-side launch
// ---------------------------------------------------------------------------
extern "C" void kernel_launch(void* const* d_in, const int* in_sizes, int n_in,
                              void* d_out, int out_size, void* d_ws, size_t ws_size,
                              hipStream_t stream) {
    const int*   ids      = (const int*)  d_in[0];
    const float* embed_W  = (const float*)d_in[1];
    const float* out_b    = (const float*)d_in[2];
    const float* ln_w     = (const float*)d_in[3];
    const float* ln_b     = (const float*)d_in[4];
    const float* norm_w   = (const float*)d_in[5];
    const float* norm_b   = (const float*)d_in[6];
    const float* inW      = (const float*)d_in[7];
    const float* convW    = (const float*)d_in[8];
    const float* convB    = (const float*)d_in[9];
    const float* xW       = (const float*)d_in[10];
    const float* dtW      = (const float*)d_in[11];
    const float* dtb      = (const float*)d_in[12];
    const float* A_log    = (const float*)d_in[13];
    const float* Dp       = (const float*)d_in[14];
    const float* outW     = (const float*)d_in[15];
    float* out = (float*)d_out;

    char* ws = (char*)d_ws;
    size_t off = 0;
    auto alloc = [&](size_t bytes) -> char* {
        char* p = ws + off;
        off += (bytes + 255) & ~(size_t)255;
        return p;
    };

    _Float16* WE   = (_Float16*)alloc((size_t)VV * DM * 2);
    _Float16* WIN  = (_Float16*)alloc((size_t)LLl * 2 * DIc * DM * 2);
    _Float16* WXP  = (_Float16*)alloc((size_t)LLl * PW * DIc * 2);   // padded 96->128
    _Float16* WDT  = (_Float16*)alloc((size_t)LLl * DIc * RRr * 2);
    _Float16* WOUT = (_Float16*)alloc((size_t)LLl * DM * DIc * 2);
    float*    X    = (float*)   alloc((size_t)BT * DM * 4);
    _Float16* HF16 = (_Float16*)alloc((size_t)BT * DM * 2);
    float*    XZ   = (float*)   alloc((size_t)BT * 2 * DIc * 4);
    float*    XC   = (float*)   alloc((size_t)BT * DIc * 4);
    _Float16* XCF  = (_Float16*)alloc((size_t)BT * DIc * 2);
    float*    PROJ = (float*)   alloc((size_t)BT * PW * 4);
    _Float16* PROJF= (_Float16*)alloc((size_t)BT * PW * 2);
    float*    DT   = (float*)   alloc((size_t)BT * DIc * 4);
    float*    Y    = (float*)   alloc((size_t)BT * DIc * 4);
    _Float16* YF16 = (_Float16*)alloc((size_t)BT * DIc * 2);
    float*    TMP  = (float*)   alloc((size_t)BT * DIc * 4);

    const int CVB = 2048;
    const int ELB = 4096;

    // weight conversion (deterministic each call)
    cvt_f16_kernel<<<CVB, 256, 0, stream>>>(embed_W, WE,  (size_t)VV * DM);
    cvt_f16_kernel<<<CVB, 256, 0, stream>>>(inW,     WIN, (size_t)LLl * 2 * DIc * DM);
    zero_f16_kernel<<<256, 256, 0, stream>>>(WXP, (size_t)LLl * PW * DIc);
    for (int l = 0; l < LLl; ++l)
        cvt_f16_kernel<<<CVB, 256, 0, stream>>>(xW + (size_t)l * 96 * DIc,
                                                WXP + (size_t)l * PW * DIc,
                                                (size_t)96 * DIc);
    cvt_f16_kernel<<<CVB, 256, 0, stream>>>(dtW,  WDT,  (size_t)LLl * DIc * RRr);
    cvt_f16_kernel<<<CVB, 256, 0, stream>>>(outW, WOUT, (size_t)LLl * DM * DIc);

    embed_kernel<<<ELB, 256, 0, stream>>>(ids, embed_W, X);

    for (int l = 0; l < LLl; ++l) {
        ln_f16_kernel<<<BT, 256, 0, stream>>>(X, norm_w + (size_t)l * DM,
                                              norm_b + (size_t)l * DM, HF16);
        {   // in_proj: (BT x DM) x (2DI x DM)^T -> XZ
            dim3 g(BT / 128, (2 * DIc) / 64);
            gemm_wmma_f16<<<g, 128, 0, stream>>>(HF16, WIN + (size_t)l * 2 * DIc * DM,
                                                 XZ, BT, 2 * DIc, DM, DM, DM, 2 * DIc);
        }
        conv_silu_kernel<<<ELB, 256, 0, stream>>>(XZ, convW + (size_t)l * DIc * KKc,
                                                  convB + (size_t)l * DIc, XC, XCF);
        {   // x_proj (padded): (BT x DI) x (PW x DI)^T -> PROJ (ldc = PW)
            dim3 g(BT / 128, PW / 64);
            gemm_wmma_f16<<<g, 128, 0, stream>>>(XCF, WXP + (size_t)l * PW * DIc,
                                                 PROJ, BT, PW, DIc, DIc, DIc, PW);
        }
        cvt_f16_kernel<<<CVB, 256, 0, stream>>>(PROJ, PROJF, (size_t)BT * PW);
        {   // dt_proj: (BT x R, lda=PW) x (DI x R)^T -> TMP
            dim3 g(BT / 128, DIc / 64);
            gemm_wmma_f16<<<g, 128, 0, stream>>>(PROJF, WDT + (size_t)l * DIc * RRr,
                                                 TMP, BT, DIc, RRr, PW, RRr, DIc);
        }
        dt_epilogue_kernel<<<ELB, 256, 0, stream>>>(TMP, dtb + (size_t)l * DIc, DT);
        scan_kernel<<<(BB * DIc) / 128, 128, 0, stream>>>(
            XC, DT, PROJ, A_log + (size_t)l * DIc * NNs, Dp + (size_t)l * DIc, Y);
        gate_kernel<<<ELB, 256, 0, stream>>>(Y, XZ, YF16);
        {   // out_proj: (BT x DI) x (DM x DI)^T -> TMP
            dim3 g(BT / 128, DM / 64);
            gemm_wmma_f16<<<g, 128, 0, stream>>>(YF16, WOUT + (size_t)l * DM * DIc,
                                                 TMP, BT, DM, DIc, DIc, DIc, DM);
        }
        add_res_kernel<<<ELB, 256, 0, stream>>>(X, TMP, (size_t)BT * DM);
    }

    ln_f16_kernel<<<BT, 256, 0, stream>>>(X, ln_w, ln_b, HF16);
    {   // vocab head: (BT x DM) x (V x DM)^T -> out
        dim3 g(BT / 128, VV / 64);
        gemm_wmma_f16<<<g, 128, 0, stream>>>(HF16, WE, out, BT, VV, DM, DM, DM, VV);
    }
    add_bias_kernel<<<ELB, 256, 0, stream>>>(out, out_b);

    (void)in_sizes; (void)n_in; (void)out_size; (void)ws_size;
}
